// EnhancedHierarchicalDeltaNet_70205535420702
// MI455X (gfx1250) — compile-verified, matrix operand order checked by reading, not run on hardware
//
#include <hip/hip_runtime.h>
#include <math.h>

// ---------------------------------------------------------------------------
// EnhancedHierarchicalDeltaNet for MI455X (gfx1250, wave32, WMMA)
// Matmuls via v_wmma_f32_16x16x32_bf16; GEMM/LDS staging via CDNA5 async
// global->LDS copies (ASYNCcnt); all fragments contiguous 2 x b128 per lane.
// ---------------------------------------------------------------------------

typedef unsigned short u16;
typedef __attribute__((ext_vector_type(16))) __bf16 v16bf;
typedef __attribute__((ext_vector_type(8)))  __bf16 v8bf;
typedef __attribute__((ext_vector_type(8)))  float  v8f;

#define B_   2
#define L_   4096
#define D_   1024
#define H_   4
#define DK_  256
#define DV_  256
#define C_   64
#define N_   64
#define BL_  (B_*L_)   // 8192
#define BH_  (B_*H_)   // 8

// ------------------------------ helpers ------------------------------------

__device__ __forceinline__ u16 f2bf(float f) {
    union { float f; unsigned u; } c; c.f = f;
    unsigned u = c.u;
    return (u16)((u + 0x7FFFu + ((u >> 16) & 1u)) >> 16);  // RNE
}
__device__ __forceinline__ float b2f(u16 h) {
    union { unsigned u; float f; } c; c.u = ((unsigned)h) << 16; return c.f;
}
__device__ __forceinline__ float sigm(float x) { return 1.f / (1.f + expf(-x)); }
__device__ __forceinline__ float siluf(float x) { return x * sigm(x); }

// CDNA5 async global->LDS copy (16B per lane), tracked by ASYNCcnt.
// VDST = LDS byte address (low 32 bits of generic LDS pointer, ISA 10.2).
__device__ __forceinline__ void async_g2l_b128(void* lds, const void* g) {
    unsigned l = (unsigned)(size_t)lds;
    asm volatile("global_load_async_to_lds_b128 %0, %1, off"
                 :: "v"(l), "v"(g) : "memory");
}
__device__ __forceinline__ void wait_async0() {
    asm volatile("s_wait_asynccnt 0" ::: "memory");
}

// 16 contiguous bf16 -> fragment (all call sites 32B aligned)
__device__ __forceinline__ v16bf ldc16(const u16* p) {
    v8bf lo = *(const v8bf*)p;
    v8bf hi = *(const v8bf*)(p + 8);
    return __builtin_shufflevector(lo, hi, 0, 1, 2, 3, 4, 5, 6, 7,
                                   8, 9, 10, 11, 12, 13, 14, 15);
}
// 16 contiguous f32 -> bf16 fragment (for the attn tile)
__device__ __forceinline__ v16bf ldc16_f32(const float* p) {
    v16bf r;
#pragma unroll
    for (int i = 0; i < 16; ++i) { union { u16 u; __bf16 b; } c; c.u = f2bf(p[i]); r[i] = c.b; }
    return r;
}
__device__ __forceinline__ v8f wmma_bf(v16bf a, v16bf b, v8f c) {
    return __builtin_amdgcn_wmma_f32_16x16x32_bf16(false, a, false, b, (short)0, c, false, false);
}
__device__ __forceinline__ float blk_sum(float v, float* red) {
    int t = threadIdx.x;
    red[t] = v; __syncthreads();
    for (int s = 128; s > 0; s >>= 1) { if (t < s) red[t] += red[t + s]; __syncthreads(); }
    float r = red[0]; __syncthreads();
    return r;
}

// ------------------------------ converts -----------------------------------

__global__ __launch_bounds__(256) void k_f2b(const float* __restrict__ src,
                                             u16* __restrict__ dst, size_t n) {
    for (size_t i = (size_t)blockIdx.x * blockDim.x + threadIdx.x; i < n;
         i += (size_t)gridDim.x * blockDim.x)
        dst[i] = f2bf(src[i]);
}

// pack transposed [Wq|Wk|Wv|Wg] -> bf16 [4096(n)][1024(k)]
__global__ __launch_bounds__(256) void k_wcatT(const float* __restrict__ Wq,
                                               const float* __restrict__ Wk,
                                               const float* __restrict__ Wv,
                                               const float* __restrict__ Wg,
                                               u16* __restrict__ dst) {
    size_t total = (size_t)4096 * 1024;
    for (size_t i = (size_t)blockIdx.x * blockDim.x + threadIdx.x; i < total;
         i += (size_t)gridDim.x * blockDim.x) {
        int nidx = (int)(i >> 10), k = (int)(i & 1023);
        int sel = nidx >> 10, c = nidx & 1023;
        const float* W = (sel == 0) ? Wq : (sel == 1) ? Wk : (sel == 2) ? Wv : Wg;
        dst[i] = f2bf(W[(size_t)k * 1024 + c]);
    }
}

// transposed convert: dst[n*K + k] = bf16(W[k*N + n])
__global__ __launch_bounds__(256) void k_wT(const float* __restrict__ W,
                                            u16* __restrict__ dst, int K, int N) {
    size_t total = (size_t)K * N;
    for (size_t i = (size_t)blockIdx.x * blockDim.x + threadIdx.x; i < total;
         i += (size_t)gridDim.x * blockDim.x) {
        int nidx = (int)(i / K), k = (int)(i % K);
        dst[i] = f2bf(W[(size_t)k * N + nidx]);
    }
}

// ------------------------------ WMMA GEMM ----------------------------------
// C[M,N] f32 = A[M,K] bf16 @ B[K,N] bf16 with B given TRANSPOSED ([N][K]).
// Block: 256 thr (8 waves), tile 64(M) x 128(N), K-step 64.
// A and B tiles staged via async global->LDS; fragments = 2 x ds_load_b128.
__global__ __launch_bounds__(256)
void k_gemm_bf16(const u16* __restrict__ A, const u16* __restrict__ BT,
                 float* __restrict__ Cm, int M, int N, int K) {
    __shared__ u16 At[64 * 64];              // [m][k]  8KB
    __shared__ u16 Bt[128 * 64];             // [n][k] 16KB
    const int tid = threadIdx.x, lane = tid & 31, wv = tid >> 5;
    const int m = lane & 15, kh = lane >> 4;
    const int bm0 = blockIdx.y * 64;
    const int bn0 = blockIdx.x * 128;
    const int tr = wv & 3;      // wave's M-tile (0..3)
    const int tcg = wv >> 2;    // wave's N-half (0..1), 4 tiles each
    const int arow = tid >> 2, aq = tid & 3;      // A staging map
    const int bcol = tid >> 1, bhalf = tid & 1;   // B staging map
    v8f acc[4] = {};
    for (int kk = 0; kk < K; kk += 64) {
        { // stage A 64x64 (async, 2 x 16B per thread)
            const u16* src = A + (size_t)(bm0 + arow) * K + kk + aq * 16;
            u16* dstp = At + arow * 64 + aq * 16;
            async_g2l_b128(dstp, src);
            async_g2l_b128(dstp + 8, src + 8);
            if (kk + 64 < K) __builtin_prefetch(src + 64, 0, 1);
        }
        { // stage B^T 128x64 (async, 4 x 16B per thread)
            const u16* src = BT + (size_t)(bn0 + bcol) * K + kk + bhalf * 32;
            u16* dstp = Bt + bcol * 64 + bhalf * 32;
            async_g2l_b128(dstp, src);
            async_g2l_b128(dstp + 8, src + 8);
            async_g2l_b128(dstp + 16, src + 16);
            async_g2l_b128(dstp + 24, src + 24);
            if (kk + 64 < K) __builtin_prefetch(src + 64, 0, 1);
        }
        wait_async0();
        __syncthreads();
#pragma unroll
        for (int ks = 0; ks < 2; ++ks) {
            v16bf af = ldc16(At + (tr * 16 + m) * 64 + ks * 32 + kh * 16);
#pragma unroll
            for (int j = 0; j < 4; ++j) {
                v16bf bf = ldc16(Bt + (tcg * 64 + j * 16 + m) * 64 + ks * 32 + kh * 16);
                acc[j] = wmma_bf(af, bf, acc[j]);
            }
        }
        __syncthreads();
    }
#pragma unroll
    for (int j = 0; j < 4; ++j) {
        int col = bn0 + tcg * 64 + j * 16 + m;
#pragma unroll
        for (int i = 0; i < 8; ++i) {
            int row = bm0 + tr * 16 + kh * 8 + i;
            Cm[(size_t)row * N + col] = acc[j][i];
        }
    }
}

// ------------------------- small projections -------------------------------
__global__ __launch_bounds__(256)
void k_smallproj(const float* __restrict__ x,
                 const float* __restrict__ Wb, const float* __restrict__ Wfd,
                 const float* __restrict__ bfd, const float* __restrict__ fdb,
                 const float* __restrict__ Wsd, const float* __restrict__ bsd,
                 const float* __restrict__ sdb, const float* __restrict__ Wfg,
                 const float* __restrict__ Wsg,
                 float* __restrict__ beta, float* __restrict__ fd,
                 float* __restrict__ sd, float* __restrict__ gF,
                 float* __restrict__ gS) {
    __shared__ float xr[1024];
    int row = blockIdx.x;                    // b*L + l
    int tid = threadIdx.x;
    for (int i = tid; i < 1024; i += 256) xr[i] = x[(size_t)row * 1024 + i];
    __syncthreads();
    if (tid < 20) {
        int which = tid >> 2, hh = tid & 3;
        const float* W = (which == 0) ? Wb : (which == 1) ? Wfd :
                         (which == 2) ? Wsd : (which == 3) ? Wfg : Wsg;
        float s = 0.f;
        for (int i = 0; i < 1024; ++i) s += xr[i] * W[i * 4 + hh];
        if (which == 1) s += bfd[hh] + fdb[hh];
        if (which == 2) s += bsd[hh] + sdb[hh];
        float sv = sigm(s);
        int b = row / L_, l = row % L_;
        size_t o = (size_t)(b * H_ + hh) * L_ + l;
        float* dst = (which == 0) ? beta : (which == 1) ? fd :
                     (which == 2) ? sd : (which == 3) ? gF : gS;
        dst[o] = sv;
    }
}

// -------------------- conv + silu + head split + l2 norm -------------------
__global__ __launch_bounds__(256)
void k_conv_heads(const float* __restrict__ qkvg,
                  const float* __restrict__ cqw, const float* __restrict__ cqb,
                  const float* __restrict__ ckw, const float* __restrict__ ckb,
                  const float* __restrict__ cvw, const float* __restrict__ cvb,
                  const float* __restrict__ beta,
                  u16* __restrict__ qb, u16* __restrict__ kbo,
                  u16* __restrict__ wbo, float* __restrict__ u4) {
    __shared__ float red[256];
    int bx = blockIdx.x;                     // (b*L + l)*H + h
    int h = bx & 3, row = bx >> 2;
    int b = row / L_, l = row % L_;
    int t = threadIdx.x;
    int ch = h * 256 + t;
    float qv = cqb[ch], kv = ckb[ch], vv = cvb[ch];
#pragma unroll
    for (int j = 0; j < 4; ++j) {
        int tt = l + j - 2;
        if (tt >= 0 && tt < L_) {
            const float* rp = qkvg + ((size_t)b * L_ + tt) * 4096;
            qv += cqw[ch * 4 + j] * rp[ch];
            kv += ckw[ch * 4 + j] * rp[1024 + ch];
            vv += cvw[ch * 4 + j] * rp[2048 + ch];
        }
    }
    qv = siluf(qv); kv = siluf(kv); vv = siluf(vv);
    float sq = blk_sum(qv * qv, red);
    float sk = blk_sum(kv * kv, red);
    float qn = qv / fmaxf(sqrtf(sq), 1e-6f);
    float kn = kv / fmaxf(sqrtf(sk), 1e-6f);
    size_t bh = (size_t)(b * H_ + h);
    float bet = beta[bh * L_ + l];
    size_t idx = (bh * L_ + l) * 256 + t;
    qb[idx] = f2bf(qn); kbo[idx] = f2bf(kn);
    wbo[idx] = f2bf(kn * bet); u4[idx] = vv * bet;
}

// ------------------------------ psi MLP ------------------------------------
__global__ __launch_bounds__(256)
void k_psi(const u16* __restrict__ kbf, const float* __restrict__ u4,
           const float* __restrict__ Wbil, const float* __restrict__ temp,
           const float* __restrict__ fw1, const float* __restrict__ fb1,
           const float* __restrict__ fw2, const float* __restrict__ fb2,
           float* __restrict__ psiArr) {
    __shared__ float red[256];
    __shared__ float fx[513];
    __shared__ float h1[128];
    int bx = blockIdx.x, bh = bx >> 6, n = bx & 63, h = bh & 3;
    size_t base = ((size_t)bh * L_ + n * 64) * 256;
    int d = threadIdx.x;
    float km = 0.f, um = 0.f, acc = 0.f;
    for (int c = 0; c < 64; ++c) {
        km += b2f(kbf[base + (size_t)c * 256 + d]);
        um += u4[base + (size_t)c * 256 + d];
    }
    km *= (1.f / 64.f); um *= (1.f / 64.f);
    const float* W = Wbil + (size_t)h * 65536;
    for (int c = 0; c < 64; ++c) {
        float s = 0.f;
        const u16* krow = kbf + base + (size_t)c * 256;
        for (int kd = 0; kd < 256; ++kd) s += b2f(krow[kd]) * W[kd * 256 + d];
        acc += s * u4[base + (size_t)c * 256 + d];
    }
    fx[d] = km; fx[256 + d] = um;
    float tot = blk_sum(acc, red);
    if (d == 0) fx[512] = tot / temp[h] * (1.f / 64.f);
    __syncthreads();
    if (d < 128) {
        float s = fb1[d];
        for (int i = 0; i < 513; ++i) s += fx[i] * fw1[i * 128 + d];
        h1[d] = siluf(s);
    }
    __syncthreads();
    float p = (d < 128) ? h1[d] * fw2[d] : 0.f;
    float s2 = blk_sum(p, red);
    if (d == 0) {
        float ps = sigm(s2 + fb2[0]);
        psiArr[bx] = fminf(fmaxf(ps, 0.01f), 0.99f);
    }
}

// per-chunk decay products gf, gs [BH, N]
__global__ __launch_bounds__(256)
void k_decay(const float* __restrict__ fd, const float* __restrict__ sd,
             float* __restrict__ gfa, float* __restrict__ gsa) {
    int i = blockIdx.x * blockDim.x + threadIdx.x;
    if (i < BH_ * N_) {
        int bh = i / N_, n = i % N_;
        const float* p = fd + (size_t)bh * L_ + n * 64;
        const float* q = sd + (size_t)bh * L_ + n * 64;
        float a = 0.f, bsum = 0.f;
        for (int c = 0; c < 64; ++c) { a += logf(p[c]); bsum += logf(q[c]); }
        gfa[i] = expf(a); gsa[i] = expf(bsum);
    }
}

// ------------------------------ chunk scan ---------------------------------
//   SbT [v][k] 128K | Uc f32 64K | UbT [v][c] 32K | Kb [c][dk] 32K |
//   KbT [dk][c] 32K | Am 16K  = 304KB (<=320KB/WGP)
#define OFF_SBT 0
#define OFF_UC  131072
#define OFF_UBT (131072 + 65536)
#define OFF_KB  (OFF_UBT + 32768)
#define OFF_KBT (OFF_KB + 32768)
#define OFF_AM  (OFF_KBT + 32768)
#define SCAN_LDS (OFF_AM + 16384)

__global__ __launch_bounds__(256)
void k_scan(const u16* __restrict__ qb, const u16* __restrict__ kb,
            const u16* __restrict__ wb, const float* __restrict__ u4,
            const float* __restrict__ psi, const float* __restrict__ gfa,
            const float* __restrict__ gsa, const float* __restrict__ gF,
            const float* __restrict__ gS,
            float* __restrict__ Sf, float* __restrict__ Ss,
            float* __restrict__ oT) {
    extern __shared__ char smem[];
    u16*   SbT = (u16*)(smem + OFF_SBT);     // [v][k] 256x256 bf16 (transposed S)
    float* Uc  = (float*)(smem + OFF_UC);    // [c][v] 64x256 f32
    u16*   UbT = (u16*)(smem + OFF_UBT);     // [v][c] 256x64 bf16 (transposed u_corr)
    u16*   Kb  = (u16*)(smem + OFF_KB);      // [c][dk] 64x256 bf16
    u16*   KbT = (u16*)(smem + OFF_KBT);     // [dk][c] 256x64 bf16
    float* Am  = (float*)(smem + OFF_AM);    // 64x64 f32

    const int tid = threadIdx.x, lane = tid & 31, wv = tid >> 5;
    const int m = lane & 15, kh = lane >> 4;
    const int bh = blockIdx.x;
    const size_t soff = (size_t)bh * DK_ * DV_;
    const size_t bhL = (size_t)bh * L_;

    for (int i = tid; i < DK_ * DV_; i += 256) { Sf[soff + i] = 0.f; Ss[soff + i] = 0.f; }
    __threadfence(); __syncthreads();

    for (int n = 0; n < N_; ++n) {
        const int l0 = n * C_;
        const size_t coff = (bhL + l0) * (size_t)DK_;
        const float psin = psi[bh * N_ + n];
        const float gfn = gfa[bh * N_ + n], gsn = gsa[bh * N_ + n];

        // (1a) Kb = k chunk (async copy) ; SbT = bf16(Sf+Ss)^T
        {
            const uint4* s4 = (const uint4*)(kb + coff);
            uint4* d4 = (uint4*)Kb;
            for (int i = tid; i < C_ * DK_ / 8; i += 256)
                async_g2l_b128(d4 + i, s4 + i);
            int v = tid;                        // one value-column per thread
            u16* drow = SbT + v * 256;
            for (int k = 0; k < 256; k += 8) {
                union { u16 t[8]; uint4 q; } pk;
#pragma unroll
                for (int j = 0; j < 8; ++j)
                    pk.t[j] = f2bf(Sf[soff + (size_t)(k + j) * 256 + v] +
                                   Ss[soff + (size_t)(k + j) * 256 + v]);
                *(uint4*)(drow + k) = pk.q;
            }
            wait_async0();
        }
        __syncthreads();

        // (1b) KbT = Kb^T  (LDS->LDS; consumed at step 10, syncs in between)
        {
            int dk = tid;
            u16* krow = KbT + dk * 64;
            for (int c = 0; c < 64; c += 8) {
                union { u16 t[8]; uint4 q; } pk;
#pragma unroll
                for (int j = 0; j < 8; ++j) pk.t[j] = Kb[(c + j) * 256 + dk];
                *(uint4*)(krow + c) = pk.q;
            }
        }

        // (2) u_adj = u - w @ S  -> Uc  (64 tiles / 8 waves)
        for (int t8 = 0; t8 < 8; ++t8) {
            int tile = wv * 8 + t8, trr = tile >> 4, tcc = tile & 15;
            v8f acc = {};
            for (int kk = 0; kk < DK_; kk += 32) {
                v16bf af = ldc16(wb + coff + (size_t)(trr * 16 + m) * DK_ + kk + kh * 16);
                v16bf bf = ldc16(SbT + (tcc * 16 + m) * 256 + kk + kh * 16);
                acc = wmma_bf(af, bf, acc);
            }
            int col = tcc * 16 + m;
#pragma unroll
            for (int i = 0; i < 8; ++i) {
                int row = trr * 16 + kh * 8 + i;
                Uc[row * DV_ + col] = u4[coff + (size_t)row * DV_ + col] - acc[i];
            }
        }

        // (3) A = tril(w @ k^T, -1) -> Am (16 tiles / 8 waves)
        for (int t2 = 0; t2 < 2; ++t2) {
            int tile = wv * 2 + t2, trr = tile >> 2, tcc = tile & 3;
            v8f acc = {};
            for (int kk = 0; kk < DK_; kk += 32) {
                v16bf af = ldc16(wb + coff + (size_t)(trr * 16 + m) * DK_ + kk + kh * 16);
                v16bf bf = ldc16(Kb + (size_t)(tcc * 16 + m) * DK_ + kk + kh * 16);
                acc = wmma_bf(af, bf, acc);
            }
            int col = tcc * 16 + m;
#pragma unroll
            for (int i = 0; i < 8; ++i) {
                int row = trr * 16 + kh * 8 + i;
                Am[row * 64 + col] = (row > col) ? acc[i] : 0.f;
            }
        }
        __syncthreads();

        // (4) forward substitution (I+A) u_corr = u_adj; column-private
        {
            int c = tid;
            for (int r = 1; r < C_; ++r) {
                float v = Uc[r * DV_ + c];
                for (int j = 0; j < r; ++j) v -= Am[r * 64 + j] * Uc[j * DV_ + c];
                Uc[r * DV_ + c] = v;
            }
            u16* urow = UbT + c * 64;
            for (int r = 0; r < C_; r += 8) {
                union { u16 t[8]; uint4 q; } pk;
#pragma unroll
                for (int j = 0; j < 8; ++j) pk.t[j] = f2bf(Uc[(r + j) * DV_ + c]);
                *(uint4*)(urow + r) = pk.q;
            }
        }
        __syncthreads();

        // (5) attn = tril(q @ k^T) -> Am (reuse)
        for (int t2 = 0; t2 < 2; ++t2) {
            int tile = wv * 2 + t2, trr = tile >> 2, tcc = tile & 3;
            v8f acc = {};
            for (int kk = 0; kk < DK_; kk += 32) {
                v16bf af = ldc16(qb + coff + (size_t)(trr * 16 + m) * DK_ + kk + kh * 16);
                v16bf bf = ldc16(Kb + (size_t)(tcc * 16 + m) * DK_ + kk + kh * 16);
                acc = wmma_bf(af, bf, acc);
            }
            int col = tcc * 16 + m;
#pragma unroll
            for (int i = 0; i < 8; ++i) {
                int row = trr * 16 + kh * 8 + i;
                Am[row * 64 + col] = (row >= col) ? acc[i] : 0.f;
            }
        }
        __syncthreads();

        // (6) o_intra = attn @ u_corr
        v8f oacc[8];
#pragma unroll
        for (int t8 = 0; t8 < 8; ++t8) {
            int tile = wv * 8 + t8, trr = tile >> 4, tcc = tile & 15;
            v8f acc = {};
#pragma unroll
            for (int kk = 0; kk < C_; kk += 32) {
                v16bf af = ldc16_f32(Am + (trr * 16 + m) * 64 + kk + kh * 16);
                v16bf bf = ldc16(UbT + (tcc * 16 + m) * 64 + kk + kh * 16);
                acc = wmma_bf(af, bf, acc);
            }
            oacc[t8] = acc;
        }
        __syncthreads();

        // (7) o += gF * (q @ Sf)
        {
            int v = tid;
            u16* drow = SbT + v * 256;
            for (int k = 0; k < 256; k += 8) {
                union { u16 t[8]; uint4 q; } pk;
#pragma unroll
                for (int j = 0; j < 8; ++j)
                    pk.t[j] = f2bf(Sf[soff + (size_t)(k + j) * 256 + v]);
                *(uint4*)(drow + k) = pk.q;
            }
        }
        __syncthreads();
#pragma unroll
        for (int t8 = 0; t8 < 8; ++t8) {
            int tile = wv * 8 + t8, trr = tile >> 4, tcc = tile & 15;
            v8f acc = {};
            for (int kk = 0; kk < DK_; kk += 32) {
                v16bf af = ldc16(qb + coff + (size_t)(trr * 16 + m) * DK_ + kk + kh * 16);
                v16bf bf = ldc16(SbT + (tcc * 16 + m) * 256 + kk + kh * 16);
                acc = wmma_bf(af, bf, acc);
            }
#pragma unroll
            for (int i = 0; i < 8; ++i) {
                int row = trr * 16 + kh * 8 + i;
                oacc[t8][i] += gF[bhL + l0 + row] * acc[i];
            }
        }
        __syncthreads();

        // (8) o += gS * (q @ Ss)
        {
            int v = tid;
            u16* drow = SbT + v * 256;
            for (int k = 0; k < 256; k += 8) {
                union { u16 t[8]; uint4 q; } pk;
#pragma unroll
                for (int j = 0; j < 8; ++j)
                    pk.t[j] = f2bf(Ss[soff + (size_t)(k + j) * 256 + v]);
                *(uint4*)(drow + k) = pk.q;
            }
        }
        __syncthreads();
#pragma unroll
        for (int t8 = 0; t8 < 8; ++t8) {
            int tile = wv * 8 + t8, trr = tile >> 4, tcc = tile & 15;
            v8f acc = {};
            for (int kk = 0; kk < DK_; kk += 32) {
                v16bf af = ldc16(qb + coff + (size_t)(trr * 16 + m) * DK_ + kk + kh * 16);
                v16bf bf = ldc16(SbT + (tcc * 16 + m) * 256 + kk + kh * 16);
                acc = wmma_bf(af, bf, acc);
            }
#pragma unroll
            for (int i = 0; i < 8; ++i) {
                int row = trr * 16 + kh * 8 + i;
                oacc[t8][i] += gS[bhL + l0 + row] * acc[i];
            }
        }
        __syncthreads();

        // (9) store chunk output
#pragma unroll
        for (int t8 = 0; t8 < 8; ++t8) {
            int tile = wv * 8 + t8, trr = tile >> 4, tcc = tile & 15;
            int col = tcc * 16 + m;
#pragma unroll
            for (int i = 0; i < 8; ++i) {
                int row = trr * 16 + kh * 8 + i;
                oT[(bhL + l0 + row) * (size_t)DV_ + col] = oacc[t8][i];
            }
        }

        // (10) kU = k^T @ u_corr, then state update (256 tiles / 8 waves)
        for (int t = 0; t < 32; ++t) {
            int tile = wv * 32 + t, trr = tile >> 4, tcc = tile & 15;
            v8f acc = {};
#pragma unroll
            for (int kk = 0; kk < C_; kk += 32) {
                v16bf af = ldc16(KbT + (trr * 16 + m) * 64 + kk + kh * 16);
                v16bf bf = ldc16(UbT + (tcc * 16 + m) * 64 + kk + kh * 16);
                acc = wmma_bf(af, bf, acc);
            }
            int col = tcc * 16 + m;
#pragma unroll
            for (int i = 0; i < 8; ++i) {
                int row = trr * 16 + kh * 8 + i;
                size_t idx = soff + (size_t)row * DV_ + col;
                float ku = acc[i];
                Sf[idx] = gfn * Sf[idx] + (1.f - psin) * ku;
                Ss[idx] = gsn * Ss[idx] + psin * ku;
            }
        }
        __threadfence();
        __syncthreads();
    }
}

// ----------------------- gated RMSNorm + bf16 pack -------------------------
__global__ __launch_bounds__(256)
void k_gate_norm(const float* __restrict__ oT, const float* __restrict__ qkvg,
                 const float* __restrict__ onw, u16* __restrict__ obf) {
    __shared__ float red[256];
    int bx = blockIdx.x;                     // (b*L + l)*H + h
    int h = bx & 3, row = bx >> 2;
    int b = row / L_, l = row % L_;
    int t = threadIdx.x;
    float o = oT[((size_t)(b * H_ + h) * L_ + l) * 256 + t];
    float ssum = blk_sum(o * o, red);
    float rms = rsqrtf(ssum / 256.f + 1e-5f);
    float g = qkvg[((size_t)b * L_ + l) * 4096 + 3072 + h * 256 + t];
    float val = o * rms * onw[t] * sigm(g);
    obf[((size_t)b * L_ + l) * 1024 + h * 256 + t] = f2bf(val);
}

// ------------------------------ host side ----------------------------------

extern "C" void kernel_launch(void* const* d_in, const int* in_sizes, int n_in,
                              void* d_out, int out_size, void* d_ws, size_t ws_size,
                              hipStream_t stream) {
    const float* x    = (const float*)d_in[0];
    const float* Wq   = (const float*)d_in[1];
    const float* Wk   = (const float*)d_in[2];
    const float* Wv   = (const float*)d_in[3];
    const float* cqw  = (const float*)d_in[4];
    const float* cqb  = (const float*)d_in[5];
    const float* ckw  = (const float*)d_in[6];
    const float* ckb  = (const float*)d_in[7];
    const float* cvw  = (const float*)d_in[8];
    const float* cvb  = (const float*)d_in[9];
    const float* Wb   = (const float*)d_in[10];
    const float* Wfd  = (const float*)d_in[11];
    const float* bfd  = (const float*)d_in[12];
    const float* fdb  = (const float*)d_in[13];
    const float* Wsd  = (const float*)d_in[14];
    const float* bsd  = (const float*)d_in[15];
    const float* sdb  = (const float*)d_in[16];
    const float* Wbil = (const float*)d_in[17];
    const float* temp = (const float*)d_in[18];
    const float* fw1  = (const float*)d_in[19];
    const float* fb1  = (const float*)d_in[20];
    const float* fw2  = (const float*)d_in[21];
    const float* fb2  = (const float*)d_in[22];
    const float* Wfg  = (const float*)d_in[23];
    const float* Wsg  = (const float*)d_in[24];
    const float* Wg   = (const float*)d_in[25];
    const float* onw  = (const float*)d_in[26];
    const float* Wo   = (const float*)d_in[27];

    char* ws = (char*)d_ws;
    size_t off = 0;
    auto alloc = [&](size_t bytes) -> char* {
        char* p = ws + off;
        off += (bytes + 255) & ~(size_t)255;
        return p;
    };
    u16*   xbf   = (u16*)alloc((size_t)BL_ * D_ * 2);
    u16*   wcatT = (u16*)alloc((size_t)D_ * 4096 * 2);
    u16*   woT   = (u16*)alloc((size_t)D_ * D_ * 2);
    float* qkvg  = (float*)alloc((size_t)BL_ * 4096 * 4);
    u16*   qbf   = (u16*)alloc((size_t)BH_ * L_ * DK_ * 2);
    u16*   kbf   = (u16*)alloc((size_t)BH_ * L_ * DK_ * 2);
    u16*   wbf   = (u16*)alloc((size_t)BH_ * L_ * DK_ * 2);
    float* u4    = (float*)alloc((size_t)BH_ * L_ * DV_ * 4);
    float* beta  = (float*)alloc((size_t)BH_ * L_ * 4);
    float* fd    = (float*)alloc((size_t)BH_ * L_ * 4);
    float* sd    = (float*)alloc((size_t)BH_ * L_ * 4);
    float* gF    = (float*)alloc((size_t)BH_ * L_ * 4);
    float* gS    = (float*)alloc((size_t)BH_ * L_ * 4);
    float* psiA  = (float*)alloc((size_t)BH_ * N_ * 4);
    float* gfa   = (float*)alloc((size_t)BH_ * N_ * 4);
    float* gsa   = (float*)alloc((size_t)BH_ * N_ * 4);
    float* Sf    = (float*)alloc((size_t)BH_ * DK_ * DV_ * 4);
    float* Ss    = (float*)alloc((size_t)BH_ * DK_ * DV_ * 4);
    float* oT    = (float*)alloc((size_t)BH_ * L_ * DV_ * 4);
    u16*   obf   = (u16*)alloc((size_t)BL_ * D_ * 2);

    k_f2b<<<4096, 256, 0, stream>>>(x, xbf, (size_t)BL_ * D_);
    k_wcatT<<<4096, 256, 0, stream>>>(Wq, Wk, Wv, Wg, wcatT);
    k_wT<<<1024, 256, 0, stream>>>(Wo, woT, D_, D_);

    dim3 g1(4096 / 128, BL_ / 64);
    k_gemm_bf16<<<g1, 256, 0, stream>>>(xbf, wcatT, qkvg, BL_, 4096, D_);

    k_smallproj<<<BL_, 256, 0, stream>>>(x, Wb, Wfd, bfd, fdb, Wsd, bsd, sdb,
                                         Wfg, Wsg, beta, fd, sd, gF, gS);
    k_conv_heads<<<BL_ * H_, 256, 0, stream>>>(qkvg, cqw, cqb, ckw, ckb, cvw, cvb,
                                               beta, qbf, kbf, wbf, u4);
    k_psi<<<BH_ * N_, 256, 0, stream>>>(kbf, u4, Wbil, temp, fw1, fb1, fw2, fb2, psiA);
    k_decay<<<2, 256, 0, stream>>>(fd, sd, gfa, gsa);

    (void)hipFuncSetAttribute(reinterpret_cast<const void*>(k_scan),
                              hipFuncAttributeMaxDynamicSharedMemorySize, SCAN_LDS);
    k_scan<<<BH_, 256, SCAN_LDS, stream>>>(qbf, kbf, wbf, u4, psiA, gfa, gsa,
                                           gF, gS, Sf, Ss, oT);

    k_gate_norm<<<BL_ * H_, 256, 0, stream>>>(oT, qkvg, onw, obf);

    dim3 g2(1024 / 128, BL_ / 64);
    k_gemm_bf16<<<g2, 256, 0, stream>>>(obf, woT, (float*)d_out, BL_, D_, D_);
}
